// MultiHeadAttention_51419348468050
// MI455X (gfx1250) — compile-verified, matrix operand order checked by reading, not run on hardware
//
#include <hip/hip_runtime.h>
#include <hip/hip_bf16.h>

typedef __bf16 bf16;
typedef bf16 v16bf __attribute__((ext_vector_type(16)));
typedef bf16 v8bf  __attribute__((ext_vector_type(8)));
typedef float v8f  __attribute__((ext_vector_type(8)));
typedef float v4f  __attribute__((ext_vector_type(4)));

#define B_     4
#define S_     2048
#define D_     768
#define H_     12
#define DEPTH_ 64

// ---------- helpers ----------

__device__ __forceinline__ bf16 f2bf(float x) {
  union { float f; unsigned u; } v; v.f = x;
  v.u += 0x7FFFu + ((v.u >> 16) & 1u);       // RNE
  union { unsigned short s; bf16 b; } o;
  o.s = (unsigned short)(v.u >> 16);
  return o.b;
}

// Load one 16x32 bf16 WMMA operand fragment for this lane.
// Per ISA 7.12.2: lane holds row (lane&15); elements are two contiguous
// 8-element chunks at K = (lane>>4)*8 and K = (lane>>4)*8 + 16.
// Caller passes p already offset by row*stride + (lane>>4)*8.
__device__ __forceinline__ v16bf load_frag(const bf16* p) {
  v8bf c0 = *(const v8bf*)(p);
  v8bf c1 = *(const v8bf*)(p + 16);
  return __builtin_shufflevector(c0, c1,
      0,1,2,3,4,5,6,7,8,9,10,11,12,13,14,15);
}

__device__ __forceinline__ v8f wmma_bf16(v16bf a, v16bf b, v8f c) {
  return __builtin_amdgcn_wmma_f32_16x16x32_bf16(
      false, a, false, b, (short)0, c, false, false);
}

// DPP16 lane-permute (stays in VALU, no LDS round trip).
template <int CTRL>
__device__ __forceinline__ float dpp_f(float x) {
  int xi = __builtin_bit_cast(int, x);
  int yi = __builtin_amdgcn_update_dpp(xi, xi, CTRL, 0xF, 0xF, false);
  return __builtin_bit_cast(float, yi);
}

// Butterfly reductions over each 16-lane row (lanes 0-15 and 16-31 separate):
// xor1 = quad_perm[1,0,3,2]=0xB1, xor2 = quad_perm[2,3,0,1]=0x4E,
// xor4 = row_half_mirror=0x141,   xor8 = row_mirror=0x140.
__device__ __forceinline__ float row16_max(float x) {
  x = fmaxf(x, dpp_f<0xB1>(x));
  x = fmaxf(x, dpp_f<0x4E>(x));
  x = fmaxf(x, dpp_f<0x141>(x));
  x = fmaxf(x, dpp_f<0x140>(x));
  return x;
}
__device__ __forceinline__ float row16_sum(float x) {
  x += dpp_f<0xB1>(x);
  x += dpp_f<0x4E>(x);
  x += dpp_f<0x141>(x);
  x += dpp_f<0x140>(x);
  return x;
}

// ---------- fp32 -> bf16 conversion (weights + activations, once) ----------

__global__ void cvt_f32_bf16(const float* __restrict__ in,
                             bf16* __restrict__ out, int n) {
  int i = (blockIdx.x * 256 + threadIdx.x) * 4;
  if (i + 3 < n) {
    v4f f = *(const v4f*)(in + i);
    out[i + 0] = f2bf(f.x);
    out[i + 1] = f2bf(f.y);
    out[i + 2] = f2bf(f.z);
    out[i + 3] = f2bf(f.w);
  }
}

// ---------- GEMM: Y = A(bf16)[M,768] @ W(bf16)[768,768]^T ----------
// MODE 0: out bf16, head-split   [B,H,S,64]
// MODE 1: out bf16, head-split-T [B,H,64,S]   (for V)
// MODE 2: out fp32 [M,768] + bias
template <int MODE>
__global__ __launch_bounds__(256) void gemm_bf16(
    const bf16* __restrict__ A, const bf16* __restrict__ W,
    const float* __restrict__ bias, void* __restrict__ out) {
  constexpr int K  = D_;
  constexpr int NK = K / 32;                   // 24 k-steps
  constexpr int LD = 40;                       // 32 + 8 pad (80B rows, 16B aligned)
  __shared__ __align__(16) bf16 As[2][128 * LD];
  __shared__ __align__(16) bf16 Ws[2][128 * LD];

  const int tid  = threadIdx.x;
  const int lane = tid & 31;
  const int wid  = tid >> 5;
  const int wm   = wid & 3;                    // 4 waves along M  (32 rows each)
  const int wn   = wid >> 2;                   // 2 waves along N  (64 cols each)
  const int m0   = blockIdx.y * 128;
  const int n0   = blockIdx.x * 128;
  const int r    = lane & 15;
  const int kb   = (lane >> 4) << 3;           // 0 or 8

  v8f zero = {0.f,0.f,0.f,0.f,0.f,0.f,0.f,0.f};
  v8f acc[2][4];
#pragma unroll
  for (int mt = 0; mt < 2; ++mt)
#pragma unroll
    for (int nt = 0; nt < 4; ++nt) acc[mt][nt] = zero;

  // stage one 128x32 bf16 tile of A and W into LDS buffer `buf`
  auto stage = [&](int kt, int buf) {
    int k0 = kt * 32;
#pragma unroll
    for (int sgm = 0; sgm < 2; ++sgm) {
      int linear = sgm * 256 + tid;
      int row = linear >> 2, cs = (linear & 3) * 8;
      *(v8bf*)&As[buf][row * LD + cs] =
          *(const v8bf*)(A + (size_t)(m0 + row) * K + k0 + cs);
      *(v8bf*)&Ws[buf][row * LD + cs] =
          *(const v8bf*)(W + (size_t)(n0 + row) * K + k0 + cs);
    }
  };

  stage(0, 0);
  __syncthreads();

  for (int kt = 0; kt < NK; ++kt) {
    const int cur = kt & 1;
    if (kt + 1 < NK) stage(kt + 1, cur ^ 1);   // prefetch next tile while computing

    v16bf afrag[2], bfrag[4];
#pragma unroll
    for (int mt = 0; mt < 2; ++mt)
      afrag[mt] = load_frag(&As[cur][(wm * 32 + mt * 16 + r) * LD + kb]);
#pragma unroll
    for (int nt = 0; nt < 4; ++nt)
      bfrag[nt] = load_frag(&Ws[cur][(wn * 64 + nt * 16 + r) * LD + kb]);
#pragma unroll
    for (int mt = 0; mt < 2; ++mt)
#pragma unroll
      for (int nt = 0; nt < 4; ++nt)
        acc[mt][nt] = wmma_bf16(afrag[mt], bfrag[nt], acc[mt][nt]);
    __syncthreads();
  }

  // store (C layout: lane = column n, vgpr i = row (lane>>4)*8 + i)
  const int lhalf = lane >> 4;
#pragma unroll
  for (int mt = 0; mt < 2; ++mt)
#pragma unroll
    for (int nt = 0; nt < 4; ++nt) {
      int n = n0 + wn * 64 + nt * 16 + r;
#pragma unroll
      for (int i = 0; i < 8; ++i) {
        int m = m0 + wm * 32 + mt * 16 + lhalf * 8 + i;
        float val = acc[mt][nt][i];
        if constexpr (MODE == 2) {
          ((float*)out)[(size_t)m * D_ + n] = val + bias[n];
        } else {
          int b = m >> 11, s = m & (S_ - 1);
          int h = n >> 6,  d = n & 63;
          if constexpr (MODE == 0)
            ((bf16*)out)[(((size_t)(b * H_ + h) * S_ + s) * DEPTH_) + d] = f2bf(val);
          else
            ((bf16*)out)[(((size_t)(b * H_ + h) * DEPTH_ + d) * S_) + s] = f2bf(val);
        }
      }
    }
}

// ---------- flash attention ----------
// grid (S/64, H, B), 128 threads (4 waves, 16 query rows each)
// Softmax runs in the exp2 domain: logits pre-scaled by (1/sqrt(64))*log2(e).
__global__ __launch_bounds__(128) void flash_attn(
    const bf16* __restrict__ Qh,   // [B,H,S,64]
    const bf16* __restrict__ Kh,   // [B,H,S,64]
    const bf16* __restrict__ Vt,   // [B,H,64,S]
    bf16* __restrict__ Oc) {       // [B,S,D] bf16 concat
  __shared__ __align__(16) bf16 Plds[4][16][72];   // per-wave P tile, C->A relayout

  const int tid   = threadIdx.x;
  const int lane  = tid & 31;
  const int wid   = tid >> 5;
  const int r     = lane & 15;
  const int lhalf = lane >> 4;
  const int kb    = lhalf * 8;
  const int h  = blockIdx.y, b = blockIdx.z;
  const int bh = b * H_ + h;
  const int q0 = blockIdx.x * 64 + wid * 16;
  const float SCALE2 = 0.125f * 1.44269504088896f;   // 1/sqrt(64) * log2(e)

  // Q fragments held in registers for the whole pass
  const bf16* Qbase = Qh + ((size_t)bh * S_ + (q0 + r)) * DEPTH_;
  v16bf qf0 = load_frag(Qbase + kb);
  v16bf qf1 = load_frag(Qbase + 32 + kb);

  const bf16* Kbase = Kh + (size_t)bh * S_ * DEPTH_;
  const bf16* Vbase = Vt + (size_t)bh * DEPTH_ * S_;

  v8f zero = {0.f,0.f,0.f,0.f,0.f,0.f,0.f,0.f};
  v8f o[4] = {zero, zero, zero, zero};
  float mrun[8], lrun[8];
#pragma unroll
  for (int i = 0; i < 8; ++i) { mrun[i] = -3.0e38f; lrun[i] = 0.f; }

  for (int kc = 0; kc < S_; kc += 64) {
    // ---- S2 = (Q K^T) * scale*log2e  for 16 q-rows x 64 keys ----
    v8f sa[4] = {zero, zero, zero, zero};
#pragma unroll
    for (int nt = 0; nt < 4; ++nt) {
      const bf16* kp = Kbase + (size_t)(kc + nt * 16 + r) * DEPTH_;
      sa[nt] = wmma_bf16(qf0, load_frag(kp + kb), sa[nt]);
      sa[nt] = wmma_bf16(qf1, load_frag(kp + 32 + kb), sa[nt]);
    }
#pragma unroll
    for (int nt = 0; nt < 4; ++nt) sa[nt] = sa[nt] * SCALE2;

    if (kc + 64 < S_) {  // pull next K/V block toward the caches
      __builtin_prefetch(Kbase + (size_t)(kc + 64 + r) * DEPTH_, 0, 1);
      __builtin_prefetch(Vbase + (size_t)r * S_ + kc + 64, 0, 1);
    }

    // ---- online softmax stats (row = lhalf*8 + i), DPP reductions ----
    float mnew[8], alpha[8], rsum[8];
#pragma unroll
    for (int i = 0; i < 8; ++i) {
      float mx = fmaxf(fmaxf(sa[0][i], sa[1][i]), fmaxf(sa[2][i], sa[3][i]));
      mx       = row16_max(mx);
      mnew[i]  = fmaxf(mrun[i], mx);
      alpha[i] = __builtin_amdgcn_exp2f(mrun[i] - mnew[i]);
      rsum[i]  = 0.f;
    }

    // ---- P = exp2(S2 - m), stage to LDS in A-layout order ----
#pragma unroll
    for (int nt = 0; nt < 4; ++nt) {
      int col = nt * 16 + r;
#pragma unroll
      for (int i = 0; i < 8; ++i) {
        float p = __builtin_amdgcn_exp2f(sa[nt][i] - mnew[i]);
        rsum[i] += p;
        Plds[wid][lhalf * 8 + i][col] = f2bf(p);
      }
    }
#pragma unroll
    for (int i = 0; i < 8; ++i) {
      lrun[i] = lrun[i] * alpha[i] + row16_sum(rsum[i]);
      mrun[i] = mnew[i];
    }
#pragma unroll
    for (int dt = 0; dt < 4; ++dt)
#pragma unroll
      for (int i = 0; i < 8; ++i) o[dt][i] *= alpha[i];

    // ---- O += P @ V  (V stored transposed: rows are d, contiguous keys) ----
#pragma unroll
    for (int j = 0; j < 2; ++j) {
      v16bf pf = load_frag(&Plds[wid][r][j * 32 + kb]);
#pragma unroll
      for (int dt = 0; dt < 4; ++dt) {
        const bf16* vp = Vbase + (size_t)(dt * 16 + r) * S_ + kc + j * 32 + kb;
        o[dt] = wmma_bf16(pf, load_frag(vp), o[dt]);
      }
    }
  }

  // ---- normalize + store bf16 concat [B,S,D] ----
#pragma unroll
  for (int i = 0; i < 8; ++i) lrun[i] = 1.f / lrun[i];
#pragma unroll
  for (int dt = 0; dt < 4; ++dt)
#pragma unroll
    for (int i = 0; i < 8; ++i) {
      int s = q0 + lhalf * 8 + i;
      int d = h * 64 + dt * 16 + r;
      Oc[((size_t)b * S_ + s) * D_ + d] = f2bf(o[dt][i] * lrun[i]);
    }
}

// ---------- launch ----------

extern "C" void kernel_launch(void* const* d_in, const int* in_sizes, int n_in,
                              void* d_out, int out_size, void* d_ws, size_t ws_size,
                              hipStream_t stream) {
  (void)in_sizes; (void)n_in; (void)out_size; (void)ws_size;
  const float* v  = (const float*)d_in[0];
  const float* k  = (const float*)d_in[1];
  const float* q  = (const float*)d_in[2];
  const float* wq = (const float*)d_in[3];
  const float* wk = (const float*)d_in[4];
  const float* wv = (const float*)d_in[5];
  const float* wo = (const float*)d_in[6];
  const float* bo = (const float*)d_in[7];
  float* out = (float*)d_out;

  char* ws = (char*)d_ws;
  size_t off = 0;
  auto alloc = [&](size_t bytes) -> void* {
    void* p = ws + off;
    off = (off + bytes + 255) & ~(size_t)255;
    return p;
  };
  const size_t WB  = (size_t)D_ * D_ * sizeof(bf16);              // weights
  const size_t AB  = (size_t)B_ * S_ * D_ * sizeof(bf16);         // activations
  const size_t QHB = (size_t)B_ * H_ * S_ * DEPTH_ * sizeof(bf16);
  bf16* wqb = (bf16*)alloc(WB);
  bf16* wkb = (bf16*)alloc(WB);
  bf16* wvb = (bf16*)alloc(WB);
  bf16* wob = (bf16*)alloc(WB);
  bf16* qb  = (bf16*)alloc(AB);
  bf16* kbf = (bf16*)alloc(AB);
  bf16* vbf = (bf16*)alloc(AB);
  bf16* Qhd = (bf16*)alloc(QHB);
  bf16* Khd = (bf16*)alloc(QHB);
  bf16* Vtd = (bf16*)alloc(QHB);
  bf16* Oc  = (bf16*)alloc(AB);

  const int wn = D_ * D_;                     // 589824 -> 576 blocks
  cvt_f32_bf16<<<wn / 1024, 256, 0, stream>>>(wq, wqb, wn);
  cvt_f32_bf16<<<wn / 1024, 256, 0, stream>>>(wk, wkb, wn);
  cvt_f32_bf16<<<wn / 1024, 256, 0, stream>>>(wv, wvb, wn);
  cvt_f32_bf16<<<wn / 1024, 256, 0, stream>>>(wo, wob, wn);
  const int an = B_ * S_ * D_;                // 6291456 -> 6144 blocks
  cvt_f32_bf16<<<an / 1024, 256, 0, stream>>>(q, qb,  an);
  cvt_f32_bf16<<<an / 1024, 256, 0, stream>>>(k, kbf, an);
  cvt_f32_bf16<<<an / 1024, 256, 0, stream>>>(v, vbf, an);

  dim3 gg(D_ / 128, (B_ * S_) / 128);         // (6, 64)
  gemm_bf16<0><<<gg, 256, 0, stream>>>(qb,  wqb, nullptr, Qhd);
  gemm_bf16<0><<<gg, 256, 0, stream>>>(kbf, wkb, nullptr, Khd);
  gemm_bf16<1><<<gg, 256, 0, stream>>>(vbf, wvb, nullptr, Vtd);

  dim3 ga(S_ / 64, H_, B_);                   // (32, 12, 4)
  flash_attn<<<ga, 128, 0, stream>>>(Qhd, Khd, Vtd, Oc);

  gemm_bf16<2><<<gg, 256, 0, stream>>>(Oc, wob, bo, out);
}
